// UNet_with_IELs_31121333027468
// MI455X (gfx1250) — matrix operand order repaired and checked by reading, not compiled, not run on hardware
//
#include <hip/hip_runtime.h>
#include <hip/hip_bf16.h>
#include <math.h>

typedef __attribute__((ext_vector_type(16))) __bf16 v16bf;
typedef __attribute__((ext_vector_type(8)))  __bf16 v8bf;
typedef __attribute__((ext_vector_type(8)))  float  v8f;

// ---------------------------------------------------------------------------
// WMMA implicit-GEMM 3x3 conv, pad=1, NCHW.
//   C[co][p] = sum_k W[co][k] * Im2col[k][p],  K = (Ca+Cb)*9, p = n*H*W + pix
// Input channels = concat of inA (Ca) and inB (Cb).
// Block: 256 threads = 8 waves; C tile = 64(M) x 128(P).
//   wave w: M-subtile (w&3)*16, P-half (w>>2)*64.
//   Per K-step: each wave loads 1 A frag + 4 B frags -> 4 v_wmma (A reuse x4).
// Staging is fully branchless: guarded loads become clamped-address loads +
// value selects (no EXEC save/restore, no divergent branches in the K loop).
// ---------------------------------------------------------------------------
__global__ __launch_bounds__(256) void conv3x3_wmma(
    const float* __restrict__ inA, int Ca,
    const float* __restrict__ inB, int Cb,
    const float* __restrict__ w,          // [Co][Ci*9] row-major
    float* __restrict__ out,              // [N][Co][H][W]
    int N, int H, int W, int Co)
{
  const int  Ci = Ca + Cb;
  const int  K  = Ci * 9;
  const long HW = (long)H * W;
  const long P  = (long)N * HW;

  __shared__ __bf16 As[64 * 32];          // [m][k]
  __shared__ __bf16 Bs[128 * 32];         // [p][k]

  const int  tid   = threadIdx.x;
  const int  lane  = tid & 31;
  const int  wave  = tid >> 5;            // 0..7
  const int  mrow  = wave & 3;            // M subtile within block
  const int  phalf = wave >> 2;           // P half (0/1)
  const int  m0    = blockIdx.x * 64;
  const long p0    = (long)blockIdx.y * 128;

  // ---- hoisted staging coordinates ----
  // A: thread -> row ms = tid>>2, contiguous k chunk of 8 at (tid&3)*8
  const int ms  = tid >> 2;
  const int kA0 = (tid & 3) * 8;
  // B: thread -> col ps = tid>>1, contiguous k chunk of 16 at (tid&1)*16
  const int  ps  = tid >> 1;
  const int  kB0 = (tid & 1) * 16;
  const long gp  = p0 + ps;
  const unsigned pok = (gp < P) ? 1u : 0u;
  long n = 0, pix = 0; int y0 = 0, x0 = 0;
  if (pok) {
    n   = gp / HW;  pix = gp - n * HW;
    y0  = (int)(pix / W);
    x0  = (int)(pix - (long)y0 * W);
  }
  const float* baseA  = inA + (n * Ca) * HW;
  const float* baseBp = inB ? (inB + (n * Cb) * HW) : inA;
  // baseB2 + ci*HW == inB + (n*Cb + ci - Ca)*HW  for ci >= Ca
  const float* baseB2 = baseBp - (long)Ca * HW;

  const bool kAligned = ((K & 3) == 0);   // float4-aligned weight rows
  const bool kMod32   = ((K & 31) == 0);

  v8f acc0 = {0.f,0.f,0.f,0.f,0.f,0.f,0.f,0.f};
  v8f acc1 = acc0, acc2 = acc0, acc3 = acc0;

  for (int k0 = 0; k0 < K; k0 += 32) {
    // ---- stage A: 64x32 weights, 8 contiguous k per thread ----
    {
      const long wbase = (long)(m0 + ms) * K + k0 + kA0;
      if (kAligned && (k0 + kA0 + 7 < K)) {
        const float4 f0 = *(const float4*)(w + wbase);
        const float4 f1 = *(const float4*)(w + wbase + 4);
        v8bf t;
        t[0] = (__bf16)f0.x; t[1] = (__bf16)f0.y;
        t[2] = (__bf16)f0.z; t[3] = (__bf16)f0.w;
        t[4] = (__bf16)f1.x; t[5] = (__bf16)f1.y;
        t[6] = (__bf16)f1.z; t[7] = (__bf16)f1.w;
        *(v8bf*)(&As[ms * 32 + kA0]) = t;
      } else {
        // branchless tail (only the inc layer, K = 27)
#pragma unroll
        for (int j = 0; j < 8; ++j) {
          int k = k0 + kA0 + j;
          unsigned ok = (k < K) ? 1u : 0u;
          float v = w[ok ? (wbase + j) : 0];
          As[ms * 32 + kA0 + j] = (__bf16)(ok ? v : 0.f);
        }
      }
    }
    // ---- stage B: 128x32 implicit im2col, branchless gather ----
    {
      const unsigned kfull = (kMod32 || (k0 + 32 <= K)) ? 1u : 0u;
#pragma unroll
      for (int j = 0; j < 16; ++j) {
        int k  = k0 + kB0 + j;
        int ci = k / 9, rs = k - ci * 9;
        int rq = rs / 3;
        int y  = y0 + rq - 1;
        int x  = x0 + (rs - rq * 3) - 1;
        unsigned ok = pok
                    & ((unsigned)y < (unsigned)H ? 1u : 0u)
                    & ((unsigned)x < (unsigned)W ? 1u : 0u)
                    & (kfull | (k < K ? 1u : 0u));
        // clamped-address load + select (no divergent branch):
        long off_real = (long)ci * HW + (long)y * W + x;
        long offi     = ok ? off_real : 0;          // !ok -> baseA[0], valid
        const float* src = (ok & (ci >= Ca ? 1u : 0u)) ? baseB2 : baseA;
        float v = src[offi];
        v = ok ? v : 0.f;
        Bs[ps * 32 + kB0 + j] = (__bf16)v;
      }
    }
    __syncthreads();

    // ---- fragments (ISA 7.12.2 16-bit layouts) ----
    const int h  = lane >> 4;
    const int ml = lane & 15;
    v16bf a;
#pragma unroll
    for (int e = 0; e < 16; ++e) {
      int vg = e >> 1, pos = e & 1;
      int ka = ((vg >> 2) * 16) + ((vg & 3) * 2) + h * 8 + pos;
      a[e] = As[(mrow * 16 + ml) * 32 + ka];
    }
    v16bf b0, b1, b2, b3;
#pragma unroll
    for (int e = 0; e < 16; ++e) {
      int kb = (e >> 3) * 16 + h * 8 + (e & 7);
      int pb = (phalf * 64 + ml) * 32 + kb;
      b0[e] = Bs[pb];
      b1[e] = Bs[pb + 16 * 32];
      b2[e] = Bs[pb + 32 * 32];
      b3[e] = Bs[pb + 48 * 32];
    }
    acc0 = __builtin_amdgcn_wmma_f32_16x16x32_bf16(false, a, false, b0, (short)0, acc0, false, false);
    acc1 = __builtin_amdgcn_wmma_f32_16x16x32_bf16(false, a, false, b1, (short)0, acc1, false, false);
    acc2 = __builtin_amdgcn_wmma_f32_16x16x32_bf16(false, a, false, b2, (short)0, acc2, false, false);
    acc3 = __builtin_amdgcn_wmma_f32_16x16x32_bf16(false, a, false, b3, (short)0, acc3, false, false);
    __syncthreads();
  }

  // ---- store C: rows r+8*h in VGPR r, col = 16*j + (lane&15) ----
  {
    const int h  = lane >> 4;
    const int ml = lane & 15;
    v8f accs[4] = {acc0, acc1, acc2, acc3};
#pragma unroll
    for (int j = 0; j < 4; ++j) {
      long col = p0 + phalf * 64 + j * 16 + ml;
      if (col < P) {
        long n2 = col / HW, pix2 = col - n2 * HW;
#pragma unroll
        for (int r = 0; r < 8; ++r) {
          int mm = m0 + mrow * 16 + r + 8 * h;
          if (mm < Co) out[(n2 * Co + mm) * HW + pix2] = accs[j][r];
        }
      }
    }
  }
}

// ---------------------------------------------------------------------------
// BatchNorm (batch statistics, as in reference) + ReLU
// ---------------------------------------------------------------------------
__global__ void bn_stats(const float* __restrict__ x, float* __restrict__ mean,
                         float* __restrict__ rstd, int C, long HW, int N)
{
  int c = blockIdx.x;
  __shared__ float ss[256], sq[256];
  float s = 0.f, q = 0.f;
  for (int n = 0; n < N; ++n) {
    const float* xp = x + ((long)n * C + c) * HW;
    for (long p = threadIdx.x; p < HW; p += blockDim.x) {
      float v = xp[p];
      s += v; q += v * v;
    }
  }
  ss[threadIdx.x] = s; sq[threadIdx.x] = q;
  __syncthreads();
  for (int st = 128; st > 0; st >>= 1) {
    if (threadIdx.x < st) { ss[threadIdx.x] += ss[threadIdx.x + st];
                            sq[threadIdx.x] += sq[threadIdx.x + st]; }
    __syncthreads();
  }
  if (threadIdx.x == 0) {
    long  tot = (long)N * HW;
    float m = ss[0] / (float)tot;
    float v = sq[0] / (float)tot - m * m;
    mean[c] = m;
    rstd[c] = rsqrtf(v + 1e-5f);
  }
}

__global__ void bn_relu(float* __restrict__ t, const float* __restrict__ mean,
                        const float* __restrict__ rstd, const float* __restrict__ g,
                        const float* __restrict__ b, int C, long HW, long total)
{
  long i = (long)blockIdx.x * blockDim.x + threadIdx.x;
  if (i >= total) return;
  long c = (i / HW) % C;
  float v = (t[i] - mean[c]) * rstd[c] * g[c] + b[c];
  t[i] = fmaxf(v, 0.f);
}

// ---------------------------------------------------------------------------
// 2x2 max-pool stride 2
// ---------------------------------------------------------------------------
__global__ void maxpool2(const float* __restrict__ x, float* __restrict__ y,
                         int N, int C, int H, int W)
{
  int Ho = H >> 1, Wo = W >> 1;
  long total = (long)N * C * Ho * Wo;
  long i = (long)blockIdx.x * blockDim.x + threadIdx.x;
  if (i >= total) return;
  int  xo = (int)(i % Wo);
  long t  = i / Wo;
  int  yo = (int)(t % Ho);
  long nc = t / Ho;                       // n*C + c
  const float* s = x + nc * (long)H * W + (long)(yo * 2) * W + xo * 2;
  float m = fmaxf(fmaxf(s[0], s[1]), fmaxf(s[W], s[W + 1]));
  y[i] = m;
}

// ---------------------------------------------------------------------------
// conv_transpose 2x2 stride 2 ('VALID') + bias;  wt: [Co][Ci][2][2]
// ---------------------------------------------------------------------------
__global__ void upconv2x2(const float* __restrict__ x, const float* __restrict__ wt,
                          const float* __restrict__ bt, float* __restrict__ y,
                          int N, int Ci, int Co, int Hi, int Wi)
{
  int Ho = 2 * Hi, Wo = 2 * Wi;
  long total = (long)N * Co * Ho * Wo;
  long i = (long)blockIdx.x * blockDim.x + threadIdx.x;
  if (i >= total) return;
  int  X = (int)(i % Wo);
  long t = i / Wo;
  int  Y = (int)(t % Ho);
  t /= Ho;
  int  o = (int)(t % Co);
  long n = t / Co;
  int yy = Y >> 1, ky = Y & 1, xx = X >> 1, kx = X & 1;
  const float* xp = x + (n * Ci) * (long)Hi * Wi + (long)yy * Wi + xx;
  const float* wp = wt + (long)o * Ci * 4 + ky * 2 + kx;
  float s = bt[o];
  long  chs = (long)Hi * Wi;
  for (int ci = 0; ci < Ci; ++ci) s += xp[(long)ci * chs] * wp[(long)ci * 4];
  y[i] = s;
}

// ---------------------------------------------------------------------------
// 1x1 output conv (Co = 2)
// ---------------------------------------------------------------------------
__global__ void conv1x1_out(const float* __restrict__ x, const float* __restrict__ w,
                            const float* __restrict__ b, float* __restrict__ y,
                            int N, int Ci, long HW)
{
  long total = (long)N * 2 * HW;
  long i = (long)blockIdx.x * blockDim.x + threadIdx.x;
  if (i >= total) return;
  long p = i % HW;
  long t = i / HW;
  int  co = (int)(t % 2);
  long n  = t / 2;
  float s = b[co];
  const float* xp = x + (n * Ci) * HW + p;
  for (int ci = 0; ci < Ci; ++ci) s += xp[(long)ci * HW] * w[co * Ci + ci];
  y[(n * 2 + co) * HW + p] = s;
}

__global__ void fillz(float* __restrict__ p, long n)
{
  long i = (long)blockIdx.x * blockDim.x + threadIdx.x;
  if (i < n) p[i] = 0.f;
}

// ---------------------------------------------------------------------------
// IELS kernels — x layout (N,2,H,W)
// ---------------------------------------------------------------------------
__global__ void iels_prep(const float* __restrict__ xo, float* __restrict__ nm,
                          float* __restrict__ dx, int N, int H, int W)
{
  long HW = (long)H * W, total = (long)N * HW;
  long i = (long)blockIdx.x * blockDim.x + threadIdx.x;
  if (i >= total) return;
  long n = i / HW, p = i - n * HW;
  int  y = (int)(p / W), x = (int)(p - (long)y * W);
  const float* c0 = xo + n * 2 * HW;
  const float* c1 = c0 + HW;
  nm[i] = (c1[p] > c0[p]) ? 1.f : 0.f;      // argmax != 0 (ties -> 0)
  float uxp = 0.f, uxm = 0.f, uyp = 0.f, uym = 0.f;
  if (x + 1 < W)  { long q = p + 1;           uxp = c1[q] - c0[q]; }
  if (x - 1 >= 0) { long q = p - 1;           uxm = c1[q] - c0[q]; }
  if (y + 1 < H)  { long q = p + W;           uyp = c1[q] - c0[q]; }
  if (y - 1 >= 0) { long q = p - W;           uym = c1[q] - c0[q]; }
  float gx = uxp - uxm, gy = uyp - uym;
  dx[i] = sqrtf(gx * gx + gy * gy + 1e-12f);
}

__global__ void iels_hsum(const float* __restrict__ in, float* __restrict__ out,
                          int N, int H, int W, int R)
{
  long HW = (long)H * W, total = (long)N * HW;
  long i = (long)blockIdx.x * blockDim.x + threadIdx.x;
  if (i >= total) return;
  long n = i / HW, p = i - n * HW;
  int  y = (int)(p / W), x = (int)(p - (long)y * W);
  const float* row = in + n * HW + (long)y * W;
  float s = 0.f;
  for (int d = -R; d <= R; ++d) {
    int xx = x + d;
    if ((unsigned)xx < (unsigned)W) s += row[xx];
  }
  out[i] = s;
}

__global__ void iels_boxv(const float* __restrict__ hs, const float* __restrict__ nm,
                          float* __restrict__ rout, int N, int H, int W)
{
  long HW = (long)H * W, total = (long)N * HW;
  long i = (long)blockIdx.x * blockDim.x + threadIdx.x;
  if (i >= total) return;
  long n = i / HW, p = i - n * HW;
  int  y = (int)(p / W), x = (int)(p - (long)y * W);
  float s = 0.f;
  for (int d = -10; d <= 10; ++d) {
    int yy = y + d;
    if ((unsigned)yy < (unsigned)H) s += hs[n * HW + (long)yy * W + x];
  }
  float box = s * (1.f / 441.f);
  rout[i] = (1.f - nm[i]) * ((box >= 0.5f) ? 1.f : 0.f);
}

__global__ void iels_update(const float* __restrict__ h7, const float* __restrict__ nm,
                            const float* __restrict__ rout, const float* __restrict__ dx,
                            float* __restrict__ trs, float* __restrict__ xio,
                            const int* __restrict__ flag, int N, int H, int W)
{
  long HW = (long)H * W, total = (long)N * HW;
  long i = (long)blockIdx.x * blockDim.x + threadIdx.x;
  if (i >= total) return;
  long n = i / HW, p = i - n * HW;
  int  y = (int)(p / W), x = (int)(p - (long)y * W);
  float s = 0.f;
  for (int d = -3; d <= 3; ++d) {
    int yy = y + d;
    if ((unsigned)yy < (unsigned)H) s += h7[n * HW + (long)yy * W + x];
  }
  float dil    = (s > 0.f) ? 1.f : 0.f;
  float region = dil * nm[i] + rout[i];
  float t      = ((trs[i] + region) > 0.f) ? 1.f : 0.f;
  trs[i] = t;
  if (flag[0] != 0) {
    xio[(n * 2 + 1) * HW + p] -= 0.1f * t * dx[i];
  }
}

// ===========================================================================
// Host side
// ===========================================================================
static const int  G_CIN [9] = {3, 64, 128, 256, 512, 1024, 512, 256, 128};
static const int  G_COUT[9] = {64, 128, 256, 512, 1024, 512, 256, 128, 64};
static const bool G_UP  [9] = {false,false,false,false,false,true,true,true,true};
// canonical index: 0=inc, 1..4=down1..4, 5..8=up1..4

struct PMap {
  int x, flag, outc_w, outc_b;
  int w1[9], g1[9], b1[9], w2[9], g2[9], b2[9], wt[9], bt[9];
};

static void build_insertion(PMap& m) {              // dict insertion order everywhere
  int i = 0;
  m.x = i++;
  for (int g = 0; g < 9; ++g) {
    m.w1[g] = i++; m.g1[g] = i++; m.b1[g] = i++;
    m.w2[g] = i++; m.g2[g] = i++; m.b2[g] = i++;
    if (G_UP[g]) { m.wt[g] = i++; m.bt[g] = i++; } else { m.wt[g] = -1; m.bt[g] = -1; }
  }
  m.outc_w = i++; m.outc_b = i++;
  m.flag = i++;
}

static void build_sorted(PMap& m, bool xFirst) {    // jax pytree sorted-key order
  int i = 0;
  if (xFirst) m.x = i++;
  const int order[5] = {1, 2, 3, 4, 0};             // down1..4, inc
  for (int t = 0; t < 5; ++t) {
    int g = order[t];
    m.b1[g] = i++; m.b2[g] = i++; m.g1[g] = i++; m.g2[g] = i++;
    m.w1[g] = i++; m.w2[g] = i++; m.wt[g] = -1; m.bt[g] = -1;
  }
  m.outc_b = i++; m.outc_w = i++;
  for (int g = 5; g < 9; ++g) {
    m.b1[g] = i++; m.b2[g] = i++; m.bt[g] = i++; m.g1[g] = i++; m.g2[g] = i++;
    m.w1[g] = i++; m.w2[g] = i++; m.wt[g] = i++;
  }
  m.flag = i++;
  if (!xFirst) m.x = i++;
}

static bool check_map(const PMap& m, const int* sz, int n_in) {
  if (n_in < 66) return false;
  bool ok = true;
  auto chk = [&](int idx, long s) {
    if (idx < 0 || idx >= n_in || (long)sz[idx] != s) ok = false;
  };
  chk(m.x, 2L * 3 * 384 * 384);
  chk(m.flag, 1);
  chk(m.outc_w, 128);
  chk(m.outc_b, 2);
  for (int g = 0; g < 9; ++g) {
    long ci = G_CIN[g], co = G_COUT[g];
    chk(m.w1[g], co * ci * 9); chk(m.g1[g], co); chk(m.b1[g], co);
    chk(m.w2[g], co * co * 9); chk(m.g2[g], co); chk(m.b2[g], co);
    if (G_UP[g]) { chk(m.wt[g], co * ci * 4); chk(m.bt[g], co); }
  }
  return ok;
}

extern "C" void kernel_launch(void* const* d_in, const int* in_sizes, int n_in,
                              void* d_out, int out_size, void* d_ws, size_t ws_size,
                              hipStream_t stream)
{
  // -------- parameter mapping (auto-detect flattening order) --------
  PMap mA, mB, mC, m;
  build_insertion(mA);
  build_sorted(mB, true);
  build_sorted(mC, false);
  if      (check_map(mA, in_sizes, n_in)) m = mA;
  else if (check_map(mB, in_sizes, n_in)) m = mB;
  else if (check_map(mC, in_sizes, n_in)) m = mC;
  else m = mA;

  auto F = [&](int i) -> const float* { return (const float*)d_in[i]; };
  const float* X    = F(m.x);
  const int*   FLAG = (const int*)d_in[m.flag];

  // -------- workspace carving --------
  char*  ws  = (char*)d_ws;
  size_t off = 0;
  auto alloc = [&](long elems) -> float* {
    float* p = (float*)(ws + off);
    off += (((size_t)elems * 4 + 255) / 256) * 256;
    return p;
  };
  float* x1   = alloc(2L * 64 * 384 * 384);
  float* x2   = alloc(2L * 128 * 192 * 192);
  float* x3   = alloc(2L * 256 * 96 * 96);
  float* x4   = alloc(2L * 512 * 48 * 48);
  float* x5   = alloc(2L * 1024 * 24 * 24);
  float* tmp  = alloc(2L * 64 * 384 * 384);   // double-conv intermediate (max)
  float* pool = alloc(2L * 64 * 192 * 192);   // max pooled tensor
  float* upb  = alloc(2L * 64 * 384 * 384);   // max transpose-conv output
  float* y4   = alloc(2L * 512 * 48 * 48);
  float* y3   = alloc(2L * 256 * 96 * 96);
  float* y2   = alloc(2L * 128 * 192 * 192);
  float* y1   = alloc(2L * 64 * 384 * 384);
  float* mean = alloc(1024);
  float* rstd = alloc(1024);
  float* nm   = alloc(2L * 384 * 384);
  float* dxb  = alloc(2L * 384 * 384);
  float* t1   = alloc(2L * 384 * 384);
  float* rout = alloc(2L * 384 * 384);
  float* trs  = alloc(2L * 384 * 384);
  (void)ws_size; (void)out_size;

  const int N = 2;

  auto launch_conv = [&](const float* inA, int Ca, const float* inB, int Cb,
                         const float* w, float* out, int H, int W, int Co) {
    long P = (long)N * H * W;
    dim3 grid((Co + 63) / 64, (unsigned)((P + 127) / 128));
    conv3x3_wmma<<<grid, 256, 0, stream>>>(inA, Ca, inB, Cb, w, out, N, H, W, Co);
  };
  auto launch_bn = [&](float* t, const float* g, const float* b, int C, int H, int W) {
    long HW = (long)H * W;
    bn_stats<<<C, 256, 0, stream>>>(t, mean, rstd, C, HW, N);
    long total = (long)N * C * HW;
    bn_relu<<<(unsigned)((total + 255) / 256), 256, 0, stream>>>(t, mean, rstd, g, b, C, HW, total);
  };
  auto dconv = [&](const float* inA, int Ca, const float* inB, int Cb,
                   int g, float* out, int H, int W) {
    int Co = G_COUT[g];
    launch_conv(inA, Ca, inB, Cb, F(m.w1[g]), tmp, H, W, Co);
    launch_bn(tmp, F(m.g1[g]), F(m.b1[g]), Co, H, W);
    launch_conv(tmp, Co, nullptr, 0, F(m.w2[g]), out, H, W, Co);
    launch_bn(out, F(m.g2[g]), F(m.b2[g]), Co, H, W);
  };
  auto launch_pool = [&](const float* in, float* out, int C, int H, int W) {
    long total = (long)N * C * (H / 2) * (W / 2);
    maxpool2<<<(unsigned)((total + 255) / 256), 256, 0, stream>>>(in, out, N, C, H, W);
  };
  auto launch_up = [&](const float* in, int g, float* out, int Ci, int Co, int Hi, int Wi) {
    long total = (long)N * Co * 2 * Hi * 2 * Wi;
    upconv2x2<<<(unsigned)((total + 255) / 256), 256, 0, stream>>>(
        in, F(m.wt[g]), F(m.bt[g]), out, N, Ci, Co, Hi, Wi);
  };

  // -------- encoder --------
  dconv(X, 3, nullptr, 0, 0, x1, 384, 384);
  launch_pool(x1, pool, 64, 384, 384);
  dconv(pool, 64, nullptr, 0, 1, x2, 192, 192);
  launch_pool(x2, pool, 128, 192, 192);
  dconv(pool, 128, nullptr, 0, 2, x3, 96, 96);
  launch_pool(x3, pool, 256, 96, 96);
  dconv(pool, 256, nullptr, 0, 3, x4, 48, 48);
  launch_pool(x4, pool, 512, 48, 48);
  dconv(pool, 512, nullptr, 0, 4, x5, 24, 24);

  // -------- decoder (concat = skip ++ upsampled, handled inside conv) --------
  launch_up(x5, 5, upb, 1024, 512, 24, 24);
  dconv(x4, 512, upb, 512, 5, y4, 48, 48);
  launch_up(y4, 6, upb, 512, 256, 48, 48);
  dconv(x3, 256, upb, 256, 6, y3, 96, 96);
  launch_up(y3, 7, upb, 256, 128, 96, 96);
  dconv(x2, 128, upb, 128, 7, y2, 192, 192);
  launch_up(y2, 8, upb, 128, 64, 192, 192);
  dconv(x1, 64, upb, 64, 8, y1, 384, 384);

  // -------- output 1x1 conv --------
  {
    long HW = 384L * 384, total = (long)N * 2 * HW;
    conv1x1_out<<<(unsigned)((total + 255) / 256), 256, 0, stream>>>(
        y1, F(m.outc_w), F(m.outc_b), (float*)d_out, N, 64, HW);
  }

  // -------- IELS: 30 iterations on d_out --------
  {
    const int H = 384, W = 384;
    long PT = (long)N * H * W;
    unsigned g1d = (unsigned)((PT + 255) / 256);
    fillz<<<g1d, 256, 0, stream>>>(trs, PT);
    float* xo = (float*)d_out;
    for (int it = 0; it < 30; ++it) {
      iels_prep  <<<g1d, 256, 0, stream>>>(xo, nm, dxb, N, H, W);
      iels_hsum  <<<g1d, 256, 0, stream>>>(nm, t1, N, H, W, 10);
      iels_boxv  <<<g1d, 256, 0, stream>>>(t1, nm, rout, N, H, W);
      iels_hsum  <<<g1d, 256, 0, stream>>>(rout, t1, N, H, W, 3);
      iels_update<<<g1d, 256, 0, stream>>>(t1, nm, rout, dxb, trs, xo, FLAG, N, H, W);
    }
  }
}
